// CrossEntropyLoss_16922171146572
// MI455X (gfx1250) — matrix operand order repaired and checked by reading, not compiled, not run on hardware
//
#include <hip/hip_runtime.h>
#include <hip/hip_bf16.h>
#include <math.h>

typedef __attribute__((ext_vector_type(16))) __bf16 v16bf;
typedef __attribute__((ext_vector_type(8)))  __bf16 v8bf;
typedef __attribute__((ext_vector_type(4)))  __bf16 v4bf;
typedef __attribute__((ext_vector_type(8)))  float  v8f;

#if __has_builtin(__builtin_amdgcn_sched_barrier)
#define SCHED_BARRIER() __builtin_amdgcn_sched_barrier(0)
#else
#define SCHED_BARRIER()
#endif

// Problem constants (reference shapes)
constexpr int S = 8192;          // tokens
constexpr int D = 2048;          // hidden
constexpr int V = 50304;         // vocab (== 393 * 128, no col masking needed)
constexpr int VSPLIT = 8;        // vocab splits for online-softmax partials
constexpr int TILES = V / 128;                            // 393
constexpr int TILES_PER = (TILES + VSPLIT - 1) / VSPLIT;  // 50

// ---------------------------------------------------------------------------
// Kernel 1: per-token label logit = dot(H[t], W[label[t]]) in fp32.
// One wave per token; float4 loads; wave shuffle reduce.
// ---------------------------------------------------------------------------
__global__ __launch_bounds__(256) void label_dot_kernel(
    const float* __restrict__ H, const float* __restrict__ W,
    const int* __restrict__ labels, float* __restrict__ out) {
  const int wave = blockIdx.x * (blockDim.x >> 5) + (threadIdx.x >> 5);
  const int lane = threadIdx.x & 31;
  const int lab = labels[wave];
  float r = 0.f;
  if (lab >= 0) {
    const float4* h4 = (const float4*)(H + (size_t)wave * D);
    const float4* w4 = (const float4*)(W + (size_t)lab * D);
#pragma unroll
    for (int j = 0; j < D / 128; ++j) {   // 16 iters: lane-strided float4
      float4 a = h4[lane + 32 * j];
      float4 b = w4[lane + 32 * j];
      r += a.x * b.x + a.y * b.y + a.z * b.z + a.w * b.w;
    }
  }
#pragma unroll
  for (int off = 1; off < 32; off <<= 1) r += __shfl_xor(r, off, 32);
  if (lane == 0) out[wave] = r;
}

// ---------------------------------------------------------------------------
// Kernel 2: WMMA GEMM (bf16 in / fp32 acc) + online softmax partials.
// Block: 256 threads = 8 waves. Tile: 128 tokens x 128 vocab per iteration.
// Register double-buffered global loads + ping-pong LDS (one barrier/K-step).
// Fragment loads batched ahead of the WMMA chain via sched_barrier so DS
// latency is pipelined (staggered s_wait_dscnt) instead of exposed per-WMMA.
// partials layout: [VSPLIT][S][2] = (running max, running sum-exp).
// ---------------------------------------------------------------------------
__global__ __launch_bounds__(256) void lse_partials_kernel(
    const float* __restrict__ H, const float* __restrict__ W,
    float* __restrict__ partials) {
  __shared__ alignas(16) __bf16 As[2][128 * 32];  // [buf][row][k], 2 x 8 KB
  __shared__ alignas(16) __bf16 Bs[2][128 * 32];  // [buf][col][k], 2 x 8 KB

  const int tid = threadIdx.x;
  const int wv = tid >> 5;
  const int lane = tid & 31;
  const int rowBase = blockIdx.x * 128;

  const int t0 = blockIdx.y * TILES_PER;
  const int t1 = (t0 + TILES_PER < TILES) ? (t0 + TILES_PER) : TILES;

  // Staging chunk mapping (same for A and B): 1024 float4s over 256 threads.
  int rs[4], cs[4];
#pragma unroll
  for (int i = 0; i < 4; ++i) {
    int idx = tid + 256 * i;
    rs[i] = idx >> 3;
    cs[i] = (idx & 7) << 2;
  }
  // A-side global base pointers are loop-invariant across vocab tiles.
  const float* pA[4];
#pragma unroll
  for (int i = 0; i < 4; ++i)
    pA[i] = H + (size_t)(rowBase + rs[i]) * D + cs[i];

  // Fragment addressing (ISA 7.12.2 bf16 layouts).
  const int mrow = lane & 15;
  const int jlo = (lane < 16) ? 0 : 1;   // A: v8bf index (K0..7 | K8..15)
  const int jb  = (lane < 16) ? 0 : 1;   // B: v8bf index (K0..7 | K8..15)

  float m_run[8], s_run[8];
#pragma unroll
  for (int i = 0; i < 8; ++i) { m_run[i] = -__builtin_inff(); s_run[i] = 0.f; }

  for (int vt = t0; vt < t1; ++vt) {
    const int colBase = vt * 128;
    const float* pB[4];
#pragma unroll
    for (int i = 0; i < 4; ++i)
      pB[i] = W + (size_t)(colBase + rs[i]) * D + cs[i];

    float4 ra[4], rb[4];
    auto issue_loads = [&](int k0) {
#pragma unroll
      for (int i = 0; i < 4; ++i) ra[i] = *(const float4*)(pA[i] + k0);
#pragma unroll
      for (int i = 0; i < 4; ++i) rb[i] = *(const float4*)(pB[i] + k0);
    };

    v8f acc[8] = {};
    issue_loads(0);  // prime the register pipeline

    for (int k0 = 0; k0 < D; k0 += 32) {
      const int buf = (k0 >> 5) & 1;

      // Convert the in-flight registers to bf16 and stage into LDS.
#pragma unroll
      for (int i = 0; i < 4; ++i) {
        v4bf b; b[0] = (__bf16)ra[i].x; b[1] = (__bf16)ra[i].y;
                b[2] = (__bf16)ra[i].z; b[3] = (__bf16)ra[i].w;
        *(v4bf*)(&As[buf][rs[i] * 32 + cs[i]]) = b;
      }
#pragma unroll
      for (int i = 0; i < 4; ++i) {
        v4bf b; b[0] = (__bf16)rb[i].x; b[1] = (__bf16)rb[i].y;
                b[2] = (__bf16)rb[i].z; b[3] = (__bf16)rb[i].w;
        *(v4bf*)(&Bs[buf][rs[i] * 32 + cs[i]]) = b;
      }
      // Kick off next K-step's global loads; their wait lands after the WMMAs.
      if (k0 + 32 < D) issue_loads(k0 + 32);

      __syncthreads();  // single barrier per step (ping-pong buffers)

      // ---- Fragment load phase: batch ALL ds_loads before any WMMA. ----
      const v8bf* arow = (const v8bf*)(&As[buf][(wv * 16 + mrow) * 32]);
      v8bf a0 = arow[jlo];
      v8bf a1 = arow[jlo + 2];
      v8bf b0[8], b1[8];
#pragma unroll
      for (int t = 0; t < 8; ++t) {
        const v8bf* bcol = (const v8bf*)(&Bs[buf][(t * 16 + mrow) * 32]);
        b0[t] = bcol[jb];
        b1[t] = bcol[jb + 2];
      }
      SCHED_BARRIER();  // keep loads above; WMMA chain below

      // ---- WMMA phase: 8 independent back-to-back matrix ops. ----
      v16bf afrag = __builtin_shufflevector(a0, a1, 0, 1, 2, 3, 4, 5, 6, 7,
                                            8, 9, 10, 11, 12, 13, 14, 15);
#pragma unroll
      for (int t = 0; t < 8; ++t) {
        v16bf bfrag = __builtin_shufflevector(b0[t], b1[t], 0, 1, 2, 3, 4, 5,
                                              6, 7, 8, 9, 10, 11, 12, 13, 14,
                                              15);
        acc[t] = __builtin_amdgcn_wmma_f32_16x16x32_bf16(
            false, afrag, false, bfrag, (short)0, acc[t], false, false);
      }
    }

    // Online softmax update over this 128-col tile.
    // C layout: VGPR slot i -> row (i + 8*(lane>=16)); lane&15 -> column.
#pragma unroll
    for (int i = 0; i < 8; ++i) {
      float mloc = acc[0][i];
#pragma unroll
      for (int t = 1; t < 8; ++t) mloc = fmaxf(mloc, acc[t][i]);
#pragma unroll
      for (int off = 1; off < 16; off <<= 1)
        mloc = fmaxf(mloc, __shfl_xor(mloc, off, 32));  // stays in 16-lane half
      float sloc = 0.f;
#pragma unroll
      for (int t = 0; t < 8; ++t) sloc += __expf(acc[t][i] - mloc);
#pragma unroll
      for (int off = 1; off < 16; off <<= 1) sloc += __shfl_xor(sloc, off, 32);
      float mnew = fmaxf(m_run[i], mloc);
      s_run[i] = s_run[i] * __expf(m_run[i] - mnew) + sloc * __expf(mloc - mnew);
      m_run[i] = mnew;
    }
  }

  // One lane per 16-lane half writes its 8 rows' (m, s).
  if ((lane & 15) == 0) {
    const int r0 = rowBase + wv * 16 + (lane >> 4) * 8;
    float* p = partials + ((size_t)blockIdx.y * S + r0) * 2;
#pragma unroll
    for (int i = 0; i < 8; ++i) { p[2 * i] = m_run[i]; p[2 * i + 1] = s_run[i]; }
  }
}

// ---------------------------------------------------------------------------
// Kernel 3: merge VSPLIT partials per token, compute NLL, deterministic
// block-level sum (no float atomics -> bit-identical across graph replays).
// ---------------------------------------------------------------------------
__global__ __launch_bounds__(256) void reduce_kernel(
    const float* __restrict__ partials, const float* __restrict__ labLogit,
    const int* __restrict__ labels, float* __restrict__ blockSums) {
  const int tok = blockIdx.x * blockDim.x + threadIdx.x;
  float m = -__builtin_inff();
#pragma unroll
  for (int v = 0; v < VSPLIT; ++v)
    m = fmaxf(m, partials[((size_t)v * S + tok) * 2]);
  float s = 0.f;
#pragma unroll
  for (int v = 0; v < VSPLIT; ++v) {
    float mv = partials[((size_t)v * S + tok) * 2];
    float sv = partials[((size_t)v * S + tok) * 2 + 1];
    s += sv * __expf(mv - m);
  }
  const int lab = labels[tok];
  float nll = (lab >= 0) ? (m + __logf(s) - labLogit[tok]) : 0.f;

  __shared__ float red[256];
  red[threadIdx.x] = nll;
  __syncthreads();
  for (int st = 128; st > 0; st >>= 1) {
    if (threadIdx.x < st) red[threadIdx.x] += red[threadIdx.x + st];
    __syncthreads();
  }
  if (threadIdx.x == 0) blockSums[blockIdx.x] = red[0];
}

__global__ void finalize_kernel(const float* __restrict__ blockSums,
                                const float* __restrict__ lw,
                                float* __restrict__ out, int n) {
  if (threadIdx.x == 0 && blockIdx.x == 0) {
    float t = 0.f;
    for (int i = 0; i < n; ++i) t += blockSums[i];
    out[0] = t * lw[0];
  }
}

// ---------------------------------------------------------------------------
// Launcher. Inputs: hidden_states [S*D] f32, head_weight [V*D] f32,
// labels [S] i32, loss_weight [1] f32. Output: 1 f32 scalar.
// Workspace: 16 + S (label logits) + VSPLIT*S*2 (partials) + 32 (block sums)
// floats ~= 560 KB.
// ---------------------------------------------------------------------------
extern "C" void kernel_launch(void* const* d_in, const int* in_sizes, int n_in,
                              void* d_out, int out_size, void* d_ws, size_t ws_size,
                              hipStream_t stream) {
  const float* H = (const float*)d_in[0];
  const float* W = (const float*)d_in[1];
  const int* labels = (const int*)d_in[2];
  const float* lw = (const float*)d_in[3];
  float* out = (float*)d_out;

  float* ws = (float*)d_ws;
  float* labLogit = ws + 16;
  float* partials = labLogit + S;                        // [VSPLIT][S][2]
  float* blockSums = partials + (size_t)VSPLIT * S * 2;  // 32 floats

  label_dot_kernel<<<S / 8, 256, 0, stream>>>(H, W, labels, labLogit);

  dim3 grid(S / 128, VSPLIT);
  lse_partials_kernel<<<grid, 256, 0, stream>>>(H, W, partials);

  reduce_kernel<<<S / 256, 256, 0, stream>>>(partials, labLogit, labels, blockSums);
  finalize_kernel<<<1, 32, 0, stream>>>(blockSums, lw, out, S / 256);
}
